// MultiHeadAttention_85435489452284
// MI455X (gfx1250) — matrix operand order repaired and checked by reading, not compiled, not run on hardware
//
#include <hip/hip_runtime.h>
#include <hip/hip_bf16.h>

// ---------------------------------------------------------------------------
// Causal MHA forward, B=4, T=2048, C=1024, H=16, Dh=64. fp32 in/out,
// internal bf16 WMMA (v_wmma_f32_16x16x32_bf16) with fp32 accumulation.
// d_out = out[B,T,C] (8,388,608 f32) ++ attn[B,H,T,T] (268,435,456 f32).
//
// MI455X: attn write (1 GB @ 23.3 TB/s ~ 43us) dominates memory traffic;
// ~138 GFLOP of GEMM work runs through bf16 WMMA. Attention = 3-pass
// (max/sum/emit) softmax with sentinel-folded causal masking: branch-free
// VALU, minimum exp count, 1.25 KB LDS per wave (high occupancy); K/V are
// L2-resident (32 MB << 192 MB) so score recompute is cheap.
// ---------------------------------------------------------------------------

typedef __bf16 bf16;
typedef __attribute__((ext_vector_type(16))) __bf16 v16bf;
typedef __attribute__((ext_vector_type(8)))  __bf16 v8bf;
typedef __attribute__((ext_vector_type(8)))  float  v8f;

#define B_   4
#define T_   2048
#define C_   1024
#define H_   16
#define DH_  64
#define M_   (B_ * T_)          // 8192 rows of x
#define SPITCH 40               // LDS staging pitch (bf16): conflict-free
#define NEGHUGE (-3.0e38f)

__device__ __forceinline__ v8f wmma_bf16(v16bf a, v16bf b, v8f c) {
  return __builtin_amdgcn_wmma_f32_16x16x32_bf16(false, a, false, b,
                                                 (short)0, c, false, false);
}

// A-frag (16x32 bf16): lane(m=l&15, half=l>>4) holds row m,
// K = {8*half+0..7, 16+8*half+0..7} -> two 16B loads.
__device__ __forceinline__ v16bf make_a_frag(v8bf lo, v8bf hi) {
  return __builtin_shufflevector(lo, hi, 0, 1, 2, 3, 4, 5, 6, 7,
                                 8, 9, 10, 11, 12, 13, 14, 15);
}
__device__ __forceinline__ v16bf load_a_frag(const bf16* __restrict__ row,
                                             int k0, int half) {
  v8bf lo = *(const v8bf*)(row + k0 + 8 * half);
  v8bf hi = *(const v8bf*)(row + k0 + 16 + 8 * half);
  return make_a_frag(lo, hi);
}
// B-frag (32x16 bf16): lane n holds column n, K = {8*half+0..7, 16+8*half+0..7}.
__device__ __forceinline__ v16bf load_b_frag(const bf16* __restrict__ row,
                                             int k0, int half) {
  v8bf lo = *(const v8bf*)(row + k0 + 8 * half);
  v8bf hi = *(const v8bf*)(row + k0 + 16 + 8 * half);
  return make_a_frag(lo, hi);
}

// --- fp32 -> bf16 ----------------------------------------------------------
__global__ void cvt_f32_bf16(const float* __restrict__ in,
                             bf16* __restrict__ out, int n) {
  int i = blockIdx.x * blockDim.x + threadIdx.x;
  int stride = gridDim.x * blockDim.x;
  for (; i < n; i += stride) out[i] = (bf16)in[i];
}

// --- NT GEMM: Y[M,N] = A[M,K] * W[N,K]^T, M=8192, N=K=1024 -----------------
// 4 waves/block; each wave: 64 rows x 64 cols (4x4 WMMA tiles, 16 wmma/K-step)
// MODE 0: bf16 q/k layout [B,H,T,Dh]; MODE 1: bf16 v-transposed [B,H,Dh,T];
// MODE 2: fp32 row-major [M,N].
template <int MODE>
__global__ void __launch_bounds__(128)
gemm_bf16_nt(const bf16* __restrict__ A, const bf16* __restrict__ W,
             bf16* __restrict__ dstb, float* __restrict__ dstf) {
  const int lane = threadIdx.x & 31;
  const int wave = threadIdx.x >> 5;
  const int n15 = lane & 15, half = lane >> 4;
  const int row0 = blockIdx.y * 256 + wave * 64;
  const int col0 = blockIdx.x * 64;

  v8f acc[4][4] = {};
  for (int k = 0; k < C_; k += 32) {
    v16bf afr[4], bfr[4];
#pragma unroll
    for (int rg = 0; rg < 4; ++rg)
      afr[rg] = load_a_frag(A + (size_t)(row0 + rg * 16 + n15) * C_, k, half);
#pragma unroll
    for (int cg = 0; cg < 4; ++cg)
      bfr[cg] = load_b_frag(W + (size_t)(col0 + cg * 16 + n15) * C_, k, half);
#pragma unroll
    for (int rg = 0; rg < 4; ++rg)
#pragma unroll
      for (int cg = 0; cg < 4; ++cg)
        acc[rg][cg] = wmma_bf16(afr[rg], bfr[cg], acc[rg][cg]);
  }
#pragma unroll
  for (int rg = 0; rg < 4; ++rg)
#pragma unroll
    for (int cg = 0; cg < 4; ++cg)
#pragma unroll
      for (int r = 0; r < 8; ++r) {
        const int grow = row0 + rg * 16 + r + 8 * half;
        const int gcol = col0 + cg * 16 + n15;
        const float v = acc[rg][cg][r];
        if (MODE == 2) {
          dstf[(size_t)grow * C_ + gcol] = v;
        } else {
          const int b = grow >> 11, t = grow & (T_ - 1);
          const int h = gcol >> 6, dh = gcol & (DH_ - 1);
          if (MODE == 0)
            dstb[((size_t)(b * H_ + h) * T_ + t) * DH_ + dh] = (bf16)v;
          else
            dstb[((size_t)(b * H_ + h) * DH_ + dh) * T_ + t] = (bf16)v;
        }
      }
}

// --- Fused attention, 3-pass softmax (max / sum / emit+PV) -----------------
// One wave per (bh, 16-row q tile). Causal mask folded into a -3e38 sentinel
// so every pass is branch-free straight-line VALU; exp(sentinel-m) == 0.
__global__ void __launch_bounds__(32)
attn_fused(const bf16* __restrict__ qb, const bf16* __restrict__ kb,
           const bf16* __restrict__ vt, float* __restrict__ attn,
           bf16* __restrict__ obuf) {
  __shared__ __align__(16) bf16 stag[16 * SPITCH];   // [16 rows][32+pad cols]
  const int lane = threadIdx.x & 31;
  const int n15 = lane & 15, half = lane >> 4;
  const int q0 = blockIdx.x * 16;
  const int bh = blockIdx.y;                          // b*16 + h
  const float scale = 0.125f;                         // 1/sqrt(64)

  const bf16* qrow = qb + ((size_t)bh * T_ + q0 + n15) * DH_;
  const v16bf qa0 = load_a_frag(qrow, 0, half);
  const v16bf qa1 = load_a_frag(qrow, 32, half);
  const bf16* kbase = kb + ((size_t)bh * T_ + n15) * DH_;

  const int ngroups = blockIdx.x + 1;                 // 16-key groups (causal)

  // ---- pass 1: row max (pure fmax chain, sentinel-masked) ----
  float mrow[8];
#pragma unroll
  for (int r = 0; r < 8; ++r) mrow[r] = NEGHUGE;

  for (int kg = 0; kg < ngroups; ++kg) {
    const bf16* krow = kbase + (size_t)kg * 16 * DH_;
    v8f s = {};
    s = wmma_bf16(qa0, load_b_frag(krow, 0, half), s);
    s = wmma_bf16(qa1, load_b_frag(krow, 32, half), s);
    const int col = kg * 16 + n15;
#pragma unroll
    for (int r = 0; r < 8; ++r) {
      const float v = (col <= q0 + r + 8 * half) ? s[r] * scale : NEGHUGE;
      mrow[r] = fmaxf(mrow[r], v);
    }
  }
#pragma unroll
  for (int mask = 1; mask <= 8; mask <<= 1)
#pragma unroll
    for (int r = 0; r < 8; ++r)
      mrow[r] = fmaxf(mrow[r], __shfl_xor(mrow[r], mask, 32));

  // ---- pass 2: row sum of exp(s - m) (masked terms underflow to 0) ----
  float lrow[8];
#pragma unroll
  for (int r = 0; r < 8; ++r) lrow[r] = 0.0f;

  for (int kg = 0; kg < ngroups; ++kg) {
    const bf16* krow = kbase + (size_t)kg * 16 * DH_;
    v8f s = {};
    s = wmma_bf16(qa0, load_b_frag(krow, 0, half), s);
    s = wmma_bf16(qa1, load_b_frag(krow, 32, half), s);
    const int col = kg * 16 + n15;
#pragma unroll
    for (int r = 0; r < 8; ++r) {
      const float v = (col <= q0 + r + 8 * half) ? s[r] * scale : NEGHUGE;
      lrow[r] += __expf(v - mrow[r]);                 // exp(NEGHUGE-m) == 0
    }
  }
#pragma unroll
  for (int mask = 1; mask <= 8; mask <<= 1)
#pragma unroll
    for (int r = 0; r < 8; ++r)
      lrow[r] += __shfl_xor(lrow[r], mask, 32);
  float inv[8];
#pragma unroll
  for (int r = 0; r < 8; ++r) inv[r] = 1.0f / lrow[r];

  // ---- pass 3: emit attn + P, accumulate O = P@V ----
  v8f o[4] = {};
  const int nch = (ngroups + 1) >> 1;                 // 32-key chunks
  for (int kc = 0; kc < nch; ++kc) {
#pragma unroll
    for (int g2 = 0; g2 < 2; ++g2) {
      const int kg = kc * 2 + g2;
      if (kg < ngroups) {                             // wave-uniform branch
        const bf16* krow = kbase + (size_t)kg * 16 * DH_;
        v8f s = {};
        s = wmma_bf16(qa0, load_b_frag(krow, 0, half), s);
        s = wmma_bf16(qa1, load_b_frag(krow, 32, half), s);
        const int col = kg * 16 + n15;
#pragma unroll
        for (int r = 0; r < 8; ++r) {
          const int row = r + 8 * half;
          const float v = (col <= q0 + row) ? s[r] * scale : NEGHUGE;
          const float p = __expf(v - mrow[r]) * inv[r];
          attn[((size_t)bh * T_ + q0 + row) * T_ + col] = p;
          stag[row * SPITCH + g2 * 16 + n15] = (bf16)p;
        }
      } else {
#pragma unroll
        for (int r = 0; r < 8; ++r)
          stag[(r + 8 * half) * SPITCH + g2 * 16 + n15] = (bf16)0.0f;
      }
    }
    // cross-lane LDS transpose handoff within the wave (DS ops are in-order)
    asm volatile("s_wait_dscnt 0x0" ::: "memory");
    v8bf lo = *(const v8bf*)(stag + n15 * SPITCH + 8 * half);
    v8bf hi = *(const v8bf*)(stag + n15 * SPITCH + 16 + 8 * half);
    const v16bf pa = make_a_frag(lo, hi);
#pragma unroll
    for (int dg = 0; dg < 4; ++dg) {
      const bf16* vrow = vt + ((size_t)bh * DH_ + dg * 16 + n15) * T_;
      o[dg] = wmma_bf16(pa, load_b_frag(vrow, kc * 32, half), o[dg]);
    }
  }

  // ---- zero-fill attn columns beyond the causal tile ----
  for (int row = 0; row < 16; ++row) {
    const size_t abase = ((size_t)bh * T_ + q0 + row) * T_;
    for (int c = q0 + 16 + lane; c < T_; c += 32) attn[abase + c] = 0.0f;
  }

  // ---- store O (bf16, [B*T, C] row-major for the output projection) ----
  const int b = bh >> 4, h = bh & 15;
#pragma unroll
  for (int dg = 0; dg < 4; ++dg)
#pragma unroll
    for (int r = 0; r < 8; ++r) {
      const int grow = q0 + r + 8 * half;
      const int gcol = h * DH_ + dg * 16 + n15;
      obuf[((size_t)b * T_ + grow) * C_ + gcol] = (bf16)o[dg][r];
    }
}

// ---------------------------------------------------------------------------
extern "C" void kernel_launch(void* const* d_in, const int* in_sizes, int n_in,
                              void* d_out, int out_size, void* d_ws, size_t ws_size,
                              hipStream_t stream) {
  const float* x  = (const float*)d_in[0];
  const float* Wq = (const float*)d_in[1];
  const float* Wk = (const float*)d_in[2];
  const float* Wv = (const float*)d_in[3];
  const float* Wo = (const float*)d_in[4];

  float* outp = (float*)d_out;                         // [B,T,C]
  float* attn = outp + (size_t)M_ * C_;                // [B,H,T,T]

  char* ws = (char*)d_ws;                              // ~92 MB total
  size_t off = 0;
  bf16* xb   = (bf16*)(ws + off); off += (size_t)M_ * C_ * 2;
  bf16* wqb  = (bf16*)(ws + off); off += (size_t)C_ * C_ * 2;
  bf16* wkb  = (bf16*)(ws + off); off += (size_t)C_ * C_ * 2;
  bf16* wvb  = (bf16*)(ws + off); off += (size_t)C_ * C_ * 2;
  bf16* wob  = (bf16*)(ws + off); off += (size_t)C_ * C_ * 2;
  bf16* qbuf = (bf16*)(ws + off); off += (size_t)M_ * C_ * 2;   // [B,H,T,Dh]
  bf16* kbuf = (bf16*)(ws + off); off += (size_t)M_ * C_ * 2;   // [B,H,T,Dh]
  bf16* vtbf = (bf16*)(ws + off); off += (size_t)M_ * C_ * 2;   // [B,H,Dh,T]
  bf16* obuf = (bf16*)(ws + off); off += (size_t)M_ * C_ * 2;   // [B*T,C]

  cvt_f32_bf16<<<2048, 256, 0, stream>>>(x,  xb,  M_ * C_);
  cvt_f32_bf16<<<1024, 256, 0, stream>>>(Wq, wqb, C_ * C_);
  cvt_f32_bf16<<<1024, 256, 0, stream>>>(Wk, wkb, C_ * C_);
  cvt_f32_bf16<<<1024, 256, 0, stream>>>(Wv, wvb, C_ * C_);
  cvt_f32_bf16<<<1024, 256, 0, stream>>>(Wo, wob, C_ * C_);

  dim3 ggrid(C_ / 64, M_ / 256), gblk(128);
  gemm_bf16_nt<0><<<ggrid, gblk, 0, stream>>>(xb, wqb, qbuf, nullptr);
  gemm_bf16_nt<0><<<ggrid, gblk, 0, stream>>>(xb, wkb, kbuf, nullptr);
  gemm_bf16_nt<1><<<ggrid, gblk, 0, stream>>>(xb, wvb, vtbf, nullptr);

  dim3 agrid(T_ / 16, B_ * H_), ablk(32);
  attn_fused<<<agrid, ablk, 0, stream>>>(qbuf, kbuf, vtbf, attn, obuf);

  gemm_bf16_nt<2><<<ggrid, gblk, 0, stream>>>(obuf, wob, nullptr, outp);

  (void)in_sizes; (void)n_in; (void)out_size; (void)ws_size;
}
